// HyperSETAttention_71614284694166
// MI455X (gfx1250) — compile-verified, matrix operand order checked by reading, not run on hardware
//
#include <hip/hip_runtime.h>
#include <math.h>

// ---------------- problem constants (from reference) ----------------
#define B_   2
#define T_   2048
#define C_   2048
#define H_   16
#define D_   128
#define BT_  (B_ * T_)      // 4096
#define BH_  (B_ * H_)      // 32
#define EPS_ 1.1920928955078125e-07f   // float32 eps (torch RMSNorm default)

#define LDSPITCH 40   // padded row pitch (elements) -> 80B: 16B aligned, conflict-free

typedef __attribute__((ext_vector_type(16))) __bf16 v16bf;
typedef __attribute__((ext_vector_type(8)))  float  v8f;

__device__ __forceinline__ unsigned short f2bf(float f) {
    unsigned u = __float_as_uint(f);
    u += 0x7FFFu + ((u >> 16) & 1u);          // round-to-nearest-even
    return (unsigned short)(u >> 16);
}

union BFrag { v16bf v; uint4 u[2]; };

// generic shared pointer -> LDS byte offset (aperture lives in high 32 bits)
__device__ __forceinline__ unsigned lds_off(const void* p) {
    return (unsigned)(unsigned long long)p;
}

// CDNA5 async DMA: 16B global -> LDS, tracked by ASYNCcnt (ISA 10. / 15.18.3 op 98)
__device__ __forceinline__ void async_copy16(const void* gptr, unsigned ldsaddr) {
    asm volatile("global_load_async_to_lds_b128 %0, %1, off"
                 :: "v"(ldsaddr), "v"((unsigned long long)gptr)
                 : "memory");
}
__device__ __forceinline__ void wait_async_le3() {
    asm volatile("s_wait_asynccnt 0x3" ::: "memory");
}
__device__ __forceinline__ void wait_async_0() {
    asm volatile("s_wait_asynccnt 0x0" ::: "memory");
}

// Read one 16x32 bf16 fragment from a padded LDS tile (ds_load_b128 x2).
// Per ISA 7.12.2: lane L holds row L%16; half=L/16 selects
// K = {8h..8h+7, 16+8h..16+8h+7} -> two contiguous 16B chunks.
__device__ __forceinline__ v16bf lds_frag(const unsigned short* tile, int rc, int half) {
    const unsigned short* p = tile + rc * LDSPITCH + half * 8;
    BFrag f;
    f.u[0] = *reinterpret_cast<const uint4*>(p);
    f.u[1] = *reinterpret_cast<const uint4*>(p + 16);
    return f.v;
}

// ---------------- generic NT bf16 GEMM:  C = alpha * A[M,K] * Bt[N,K]^T ----
// block = 256 threads (8 waves, 2x4), block tile 64(M) x 128(N), k-step 32.
// Double-buffered async-to-LDS staging; wave tile 32x32 = 2x2 WMMA fragments.
// Requires M%64==0, N%128==0, K%32==0 (true for all call sites here).
__global__ __launch_bounds__(256)
void gemm_nt_bf16(const unsigned short* __restrict__ A,
                  const unsigned short* __restrict__ Bt,
                  float* __restrict__ C,
                  int M, int N, int K, int lda, int ldb, int ldc,
                  long long sA, long long sB, long long sC, float alpha) {
    __shared__ unsigned short Al[2][64  * LDSPITCH];
    __shared__ unsigned short Bl[2][128 * LDSPITCH];

    const int z = blockIdx.z;
    A  += (long long)z * sA;
    Bt += (long long)z * sB;
    C  += (long long)z * sC;

    const int tid  = threadIdx.x;
    const int wave = tid >> 5;
    const int lane = tid & 31;
    const int half = lane >> 4;
    const int l16  = lane & 15;
    const int wm   = wave >> 2;          // 0..1
    const int wn   = wave & 3;           // 0..3
    const int row0 = blockIdx.y * 64;    // block tile origin
    const int col0 = blockIdx.x * 128;

    // cooperative 16B-chunk assignment for staging (per thread, per stage):
    // A tile 64x32 bf16 = 4KB = 256 chunks; B tile 128x32 = 8KB = 512 chunks
    const int ar  = tid >> 2;            // A row 0..63
    const int ac  = (tid & 3) * 8;       // A k-offset {0,8,16,24}
    const int br0 = tid >> 2;            // B rows 0..63   (chunk tid)
    const int br1 = (tid + 256) >> 2;    // B rows 64..127 (chunk tid+256)
    const int bc  = (tid & 3) * 8;

    v8f acc[2][2] = {};

    const int steps = K >> 5;

    // prologue: stage k-slab 0 into buffer 0
    {
        async_copy16(A  + (long long)(row0 + ar)  * lda + ac,
                     lds_off(&Al[0][ar  * LDSPITCH + ac]));
        async_copy16(Bt + (long long)(col0 + br0) * ldb + bc,
                     lds_off(&Bl[0][br0 * LDSPITCH + bc]));
        async_copy16(Bt + (long long)(col0 + br1) * ldb + bc,
                     lds_off(&Bl[0][br1 * LDSPITCH + bc]));
    }

    for (int s = 0; s < steps; ++s) {
        const int cur = s & 1;
        if (s + 1 < steps) {
            const int nxt = cur ^ 1;
            const int k1  = (s + 1) << 5;
            async_copy16(A  + (long long)(row0 + ar)  * lda + k1 + ac,
                         lds_off(&Al[nxt][ar  * LDSPITCH + ac]));
            async_copy16(Bt + (long long)(col0 + br0) * ldb + k1 + bc,
                         lds_off(&Bl[nxt][br0 * LDSPITCH + bc]));
            async_copy16(Bt + (long long)(col0 + br1) * ldb + k1 + bc,
                         lds_off(&Bl[nxt][br1 * LDSPITCH + bc]));
            wait_async_le3();   // stage s landed; stage s+1 still in flight
        } else {
            wait_async_0();     // final stage landed
        }
        __syncthreads();        // collective: every wave's DMA for stage s done

        v16bf a0 = lds_frag(Al[cur], wm * 32 + l16,      half);
        v16bf a1 = lds_frag(Al[cur], wm * 32 + 16 + l16, half);
        v16bf b0 = lds_frag(Bl[cur], wn * 32 + l16,      half);
        v16bf b1 = lds_frag(Bl[cur], wn * 32 + 16 + l16, half);

        acc[0][0] = __builtin_amdgcn_wmma_f32_16x16x32_bf16(false, a0, false, b0,
                        (short)0, acc[0][0], false, false);
        acc[0][1] = __builtin_amdgcn_wmma_f32_16x16x32_bf16(false, a0, false, b1,
                        (short)0, acc[0][1], false, false);
        acc[1][0] = __builtin_amdgcn_wmma_f32_16x16x32_bf16(false, a1, false, b0,
                        (short)0, acc[1][0], false, false);
        acc[1][1] = __builtin_amdgcn_wmma_f32_16x16x32_bf16(false, a1, false, b1,
                        (short)0, acc[1][1], false, false);

        __syncthreads();        // all waves done reading buf before it is re-staged
    }

    // C/D layout: VGPR r -> M = r + 8*half, N = lane%16 (ISA 7.12.2)
    const int rowBase = row0 + wm * 32;
    const int colBase = col0 + wn * 32;
    #pragma unroll
    for (int fm = 0; fm < 2; ++fm)
        #pragma unroll
        for (int fn = 0; fn < 2; ++fn) {
            const int col = colBase + fn * 16 + l16;
            #pragma unroll
            for (int r = 0; r < 8; ++r) {
                const int row = rowBase + fm * 16 + r + 8 * half;
                C[(long long)row * ldc + col] = acc[fm][fn][r] * alpha;
            }
        }
}

// ---------------- elementwise f32 -> bf16 ----------------
__global__ void f32_to_bf16(const float* __restrict__ src,
                            unsigned short* __restrict__ dst, long long n) {
    long long i = (long long)blockIdx.x * blockDim.x + threadIdx.x;
    long long stride = (long long)gridDim.x * blockDim.x;
    for (; i < n; i += stride) dst[i] = f2bf(src[i]);
}

// ---------------- W^T (bf16) via LDS tile ----------------
__global__ __launch_bounds__(256)
void transpose_convert(const float* __restrict__ W, unsigned short* __restrict__ Wt) {
    __shared__ unsigned short tile[32][33];
    const int bx = blockIdx.x * 32, by = blockIdx.y * 32;
    const int tx = threadIdx.x & 31, tr = threadIdx.x >> 5;  // 8 rows/pass
    for (int r = tr; r < 32; r += 8)
        tile[r][tx] = f2bf(W[(long long)(by + r) * C_ + bx + tx]);
    __syncthreads();
    for (int r = tr; r < 32; r += 8)
        Wt[(long long)(bx + r) * C_ + by + tx] = tile[tx][r];
}

// ---------------- RMS norm -> q,k (row major bf16) and v^T (bf16) ----------
// one wave per (b, t, h); D = 128 = 32 lanes * 4
__global__ __launch_bounds__(256)
void rmsnorm_qkv(const float* __restrict__ w,
                 const float* __restrict__ g1, const float* __restrict__ g2,
                 const float* __restrict__ g3,
                 unsigned short* __restrict__ q, unsigned short* __restrict__ k,
                 unsigned short* __restrict__ vT) {
    const int gw = blockIdx.x * 8 + (threadIdx.x >> 5);
    if (gw >= BT_ * H_) return;
    const int lane = threadIdx.x & 31;
    const int bt = gw / H_, h = gw % H_;
    const int d0 = lane * 4;

    const float* wp = w + (long long)bt * C_ + h * D_;
    float4 wv = *reinterpret_cast<const float4*>(wp + d0);
    float ss = wv.x * wv.x + wv.y * wv.y + wv.z * wv.z + wv.w * wv.w;
    #pragma unroll
    for (int off = 16; off > 0; off >>= 1) ss += __shfl_xor(ss, off, 32);
    const float rstd = rsqrtf(ss * (1.0f / D_) + EPS_);

    const int b = bt / T_, t = bt % T_;
    const int bh = b * H_ + h;
    const float n0 = wv.x * rstd, n1 = wv.y * rstd, n2 = wv.z * rstd, n3 = wv.w * rstd;

    float4 gv1 = *reinterpret_cast<const float4*>(g1 + d0);
    float4 gv2 = *reinterpret_cast<const float4*>(g2 + d0);
    float4 gv3 = *reinterpret_cast<const float4*>(g3 + d0);

    const long long qoff = ((long long)bh * T_ + t) * D_ + d0;
    *reinterpret_cast<ushort4*>(q + qoff) =
        make_ushort4(f2bf(n0 * gv1.x), f2bf(n1 * gv1.y), f2bf(n2 * gv1.z), f2bf(n3 * gv1.w));
    *reinterpret_cast<ushort4*>(k + qoff) =
        make_ushort4(f2bf(n0 * gv2.x), f2bf(n1 * gv2.y), f2bf(n2 * gv2.z), f2bf(n3 * gv2.w));

    const long long vb = (long long)bh * D_ * T_ + (long long)d0 * T_ + t;
    vT[vb]             = f2bf(n0 * gv3.x);
    vT[vb + T_]        = f2bf(n1 * gv3.y);
    vT[vb + 2LL * T_]  = f2bf(n2 * gv3.z);
    vT[vb + 3LL * T_]  = f2bf(n3 * gv3.w);
}

// ---------------- per-row softmax stats (max, 1/sumexp) ----------------
__global__ __launch_bounds__(256)
void row_stats(const float* __restrict__ att,
               float* __restrict__ rmax, float* __restrict__ rinv) {
    __shared__ float red[256];
    const int bh = blockIdx.y, i = blockIdx.x, tid = threadIdx.x;
    const float* row = att + ((long long)bh * T_ + i) * T_;

    float m = -INFINITY;
    for (int j = tid; j < T_; j += 256) m = fmaxf(m, row[j]);
    red[tid] = m; __syncthreads();
    for (int s = 128; s > 0; s >>= 1) {
        if (tid < s) red[tid] = fmaxf(red[tid], red[tid + s]);
        __syncthreads();
    }
    m = red[0]; __syncthreads();

    float sum = 0.f;
    for (int j = tid; j < T_; j += 256) sum += __expf(row[j] - m);
    red[tid] = sum; __syncthreads();
    for (int s = 128; s > 0; s >>= 1) {
        if (tid < s) red[tid] += red[tid + s];
        __syncthreads();
    }
    if (tid == 0) {
        rmax[(long long)bh * T_ + i] = m;
        rinv[(long long)bh * T_ + i] = 1.0f / red[0];
    }
}

// ---------------- per-column online softmax stats (coalesced) --------------
__global__ __launch_bounds__(256)
void col_stats(const float* __restrict__ att,
               float* __restrict__ cmax, float* __restrict__ cinv) {
    const int bh = blockIdx.y;
    const int j = blockIdx.x * 256 + threadIdx.x;
    const float* base = att + (long long)bh * T_ * T_ + j;
    float m = -INFINITY, s = 0.f;
    for (int i = 0; i < T_; ++i) {
        const float v = base[(long long)i * T_];
        const float nm = fmaxf(m, v);
        s = s * __expf(m - nm) + __expf(v - nm);
        m = nm;
    }
    cmax[(long long)bh * T_ + j] = m;
    cinv[(long long)bh * T_ + j] = 1.0f / s;
}

// ---------------- fused row+col softmax -> bf16 score ----------------
__global__ void score_kernel(const float* __restrict__ att,
                             const float* __restrict__ rmax, const float* __restrict__ rinv,
                             const float* __restrict__ cmax, const float* __restrict__ cinv,
                             unsigned short* __restrict__ score) {
    const long long total = (long long)BH_ * T_ * T_;
    long long idx = (long long)blockIdx.x * blockDim.x + threadIdx.x;
    const long long stride = (long long)gridDim.x * blockDim.x;
    for (; idx < total; idx += stride) {
        const long long bh  = idx / ((long long)T_ * T_);
        const long long rem = idx - bh * (long long)T_ * T_;
        const int i = (int)(rem / T_);
        const int j = (int)(rem - (long long)i * T_);
        const float a = att[idx];
        const float sr = __expf(a - rmax[bh * T_ + i]) * rinv[bh * T_ + i];
        const float sc = __expf(a - cmax[bh * T_ + j]) * cinv[bh * T_ + j];
        score[idx] = f2bf(sr + sc);
    }
}

// ---------------- y[bh][t][d] (f32) -> y[b*T+t][h*D+d] (bf16) --------------
__global__ void reorder_y(const float* __restrict__ ybh, unsigned short* __restrict__ ybf) {
    const long long total = (long long)BH_ * T_ * D_;
    long long idx = (long long)blockIdx.x * blockDim.x + threadIdx.x;
    const long long stride = (long long)gridDim.x * blockDim.x;
    for (; idx < total; idx += stride) {
        const int d = (int)(idx % D_);
        const long long r = idx / D_;
        const int t = (int)(r % T_);
        const int bh = (int)(r / T_);
        const int b = bh / H_, h = bh % H_;
        ybf[((long long)(b * T_ + t)) * C_ + h * D_ + d] = f2bf(ybh[idx]);
    }
}

// ---------------- host side ----------------
extern "C" void kernel_launch(void* const* d_in, const int* in_sizes, int n_in,
                              void* d_out, int out_size, void* d_ws, size_t ws_size,
                              hipStream_t stream) {
    const float* x  = (const float*)d_in[0];
    const float* W  = (const float*)d_in[1];
    const float* g1 = (const float*)d_in[2];
    const float* g2 = (const float*)d_in[3];
    const float* g3 = (const float*)d_in[4];
    float* out = (float*)d_out;

    char* ws = (char*)d_ws;
    auto alloc = [&](size_t bytes) -> char* {
        char* p = ws;
        ws += (bytes + 255) & ~(size_t)255;
        return p;
    };

    unsigned short* xbf  = (unsigned short*)alloc((size_t)BT_ * C_ * 2);
    unsigned short* Wbf  = (unsigned short*)alloc((size_t)C_ * C_ * 2);
    unsigned short* Wtbf = (unsigned short*)alloc((size_t)C_ * C_ * 2);
    float*          wbuf = (float*)         alloc((size_t)BT_ * C_ * 4);
    unsigned short* qb   = (unsigned short*)alloc((size_t)BH_ * T_ * D_ * 2);
    unsigned short* kb   = (unsigned short*)alloc((size_t)BH_ * T_ * D_ * 2);
    unsigned short* vT   = (unsigned short*)alloc((size_t)BH_ * D_ * T_ * 2);
    float*          att  = (float*)         alloc((size_t)BH_ * T_ * T_ * 4);
    float*          rmax = (float*)         alloc((size_t)BH_ * T_ * 4);
    float*          rinv = (float*)         alloc((size_t)BH_ * T_ * 4);
    float*          cmax = (float*)         alloc((size_t)BH_ * T_ * 4);
    float*          cinv = (float*)         alloc((size_t)BH_ * T_ * 4);
    unsigned short* scr  = (unsigned short*)alloc((size_t)BH_ * T_ * T_ * 2);
    float*          ybh  = (float*)         alloc((size_t)BH_ * T_ * D_ * 4);
    unsigned short* ybf  = (unsigned short*)alloc((size_t)BT_ * C_ * 2);

    // 1) precision prep
    f32_to_bf16<<<2048, 256, 0, stream>>>(x, xbf, (long long)BT_ * C_);
    f32_to_bf16<<<2048, 256, 0, stream>>>(W, Wbf, (long long)C_ * C_);
    transpose_convert<<<dim3(C_ / 32, C_ / 32), 256, 0, stream>>>(W, Wtbf);

    // 2) w = x @ W^T     (M=4096, N=2048, K=2048)
    gemm_nt_bf16<<<dim3(C_ / 128, BT_ / 64, 1), 256, 0, stream>>>(
        xbf, Wbf, wbuf, BT_, C_, C_, C_, C_, C_, 0, 0, 0, 1.0f);

    // 3) RMS norm -> q, k, v^T (bf16)
    rmsnorm_qkv<<<(BT_ * H_) / 8, 256, 0, stream>>>(wbuf, g1, g2, g3, qb, kb, vT);

    // 4) att = q k^T / sqrt(D)   per (b,h): M=N=2048, K=128
    gemm_nt_bf16<<<dim3(T_ / 128, T_ / 64, BH_), 256, 0, stream>>>(
        qb, kb, att, T_, T_, D_, D_, D_, T_,
        (long long)T_ * D_, (long long)T_ * D_, (long long)T_ * T_,
        0.08838834764831845f /* 1/sqrt(128) */);

    // 5) softmax stats (rows + columns), fused score in bf16
    row_stats<<<dim3(T_, BH_), 256, 0, stream>>>(att, rmax, rinv);
    col_stats<<<dim3(T_ / 256, BH_), 256, 0, stream>>>(att, cmax, cinv);
    score_kernel<<<8192, 256, 0, stream>>>(att, rmax, rinv, cmax, cinv, scr);

    // 6) y = score @ v   per (b,h): M=2048, N=128, K=2048  (v stored transposed)
    gemm_nt_bf16<<<dim3(D_ / 128, T_ / 64, BH_), 256, 0, stream>>>(
        scr, vT, ybh, T_, D_, T_, T_, T_, D_,
        (long long)T_ * T_, (long long)D_ * T_, (long long)T_ * D_, 1.0f);

    // 7) reorder bhtd -> (b t)(h d), convert to bf16
    reorder_y<<<8192, 256, 0, stream>>>(ybh, ybf);

    // 8) out = y @ W     (M=4096, N=2048, K=2048; Bt = W^T)
    gemm_nt_bf16<<<dim3(C_ / 128, BT_ / 64, 1), 256, 0, stream>>>(
        ybf, Wtbf, out, BT_, C_, C_, C_, C_, C_, 0, 0, 0, 1.0f);
}
